// MEATransformerBlock_83081847374479
// MI455X (gfx1250) — matrix operand order repaired and checked by reading, not compiled, not run on hardware
//
#include <hip/hip_runtime.h>
#include <hip/hip_bf16.h>
#include <math.h>

// ---------------- types & helpers ----------------
typedef __attribute__((ext_vector_type(16))) __bf16 bf16x16;
typedef __attribute__((ext_vector_type(8)))  float  f32x8;
typedef __attribute__((__vector_size__(16))) int    vs4i;   // matches async-LDS builtin param type

union FragU { bf16x16 v; uint4 q[2]; unsigned short u[16]; };
union U4    { uint4 q; unsigned short s[8]; };

#define LDSP 40   // LDS row pitch in ushorts (80B rows -> 16B aligned b128 slots)

#if defined(__has_builtin)
#if __has_builtin(__builtin_amdgcn_global_load_async_to_lds_b128)
#define HAVE_ASYNC_LDS 1
#endif
#endif

#define AS1G __attribute__((address_space(1)))
#define AS3L __attribute__((address_space(3)))

__device__ inline void async_wait0() {
#ifdef HAVE_ASYNC_LDS
#if __has_builtin(__builtin_amdgcn_s_wait_asynccnt)
    __builtin_amdgcn_s_wait_asynccnt(0);
#else
    asm volatile("s_wait_asynccnt 0x0" ::: "memory");
#endif
#endif
}

__device__ inline unsigned short f32_to_bf16b(float f) {
    unsigned u = __builtin_bit_cast(unsigned, f);
    unsigned r = u + 0x7FFFu + ((u >> 16) & 1u);
    return (unsigned short)(r >> 16);
}
__device__ inline float gelu_f(float x) {
    return 0.5f * x * (1.0f + erff(x * 0.70710678118654752440f));
}

// copy 8 bf16 (16B) global -> LDS, async when available
__device__ inline void stage8(const unsigned short* __restrict__ src, unsigned short* dst) {
#ifdef HAVE_ASYNC_LDS
    __builtin_amdgcn_global_load_async_to_lds_b128(
        (AS1G vs4i*)const_cast<unsigned short*>(src),
        (AS3L vs4i*)dst, 0, 0);
#else
    *(uint4*)dst = *(const uint4*)src;
#endif
}

// A fragment: lane holds row m = base + (lane&15); K = {0..7,16..23} (lanes<16) or {8..15,24..31}
__device__ inline void load_fragA(FragU& f, const unsigned short* As, int mbase, int lane) {
    int m   = mbase + (lane & 15);
    int klo = (lane < 16) ? 0 : 8;
    f.q[0] = *(const uint4*)(As + m * LDSP + klo);
    f.q[1] = *(const uint4*)(As + m * LDSP + klo + 16);
}
// B fragment from LDS [n][k]: lane holds col n = base + (lane&15); K = {0..7,16..23} (lanes<16) or {8..15,24..31}
__device__ inline void load_fragB(FragU& f, const unsigned short* Bs, int nbase, int lane) {
    int n   = nbase + (lane & 15);
    int klo = (lane < 16) ? 0 : 8;
    f.q[0] = *(const uint4*)(Bs + n * LDSP + klo);
    f.q[1] = *(const uint4*)(Bs + n * LDSP + klo + 16);
}
__device__ inline f32x8 wmma_bf16(const FragU& a, const FragU& b, f32x8 c) {
    return __builtin_amdgcn_wmma_f32_16x16x32_bf16(false, a.v, false, b.v, (short)0, c, false, false);
}

// ---------------- generic GEMM: C = A[MxK] * Bt^T (+bias)(+residual) ----------------
// A row-major [M,K]; Bt row-major [N,K] (pre-transposed) -> both tiles stage as direct copies.
// Double-buffered LDS, async global->LDS pipeline. Optional per-batch Bt (BM | rowsPerBatch).
template<int BM, int WR, int WC>
__global__ __launch_bounds__(256) void gemm_bf16_wmma(
    const unsigned short* __restrict__ A, int lda,
    const unsigned short* __restrict__ Bt, int ldb,
    long long bBatchStride, int rowsPerBatch,
    float* __restrict__ C, unsigned short* __restrict__ Cbf, int ldc,
    const float* __restrict__ bias, const float* __restrict__ residual,
    int M, int N, int K)
{
    static_assert(WR * WC == 8, "8 waves");
    constexpr int BN  = 64;
    constexpr int WTM = BM / WR;
    constexpr int WTN = BN / WC;
    constexpr int AM  = WTM / 16;
    constexpr int AN  = WTN / 16;

    __shared__ unsigned short As[2][BM * LDSP];
    __shared__ unsigned short Bs[2][BN * LDSP];

    const int tid  = threadIdx.x;
    const int wid  = tid >> 5;
    const int lane = tid & 31;
    const int waveM = (wid % WR) * WTM;
    const int waveN = (wid / WR) * WTN;
    const int tileM = blockIdx.x * BM;
    const int tileN = blockIdx.y * BN;

    const unsigned short* Bbase = Bt;
    if (bBatchStride != 0)
        Bbase += (long long)(tileM / rowsPerBatch) * bBatchStride;

    auto stage = [&](int buf, int k0) {
        constexpr int VECS = (BM * 32) / (256 * 8);
#pragma unroll
        for (int v = 0; v < VECS; ++v) {          // A tile: BM x 32 direct rows
            int idx  = tid + v * 256;
            int row  = idx >> 2;
            int col8 = (idx & 3) * 8;
            unsigned short* dst = &As[buf][row * LDSP + col8];
            int gm = tileM + row;
            if (gm < M) stage8(A + (long long)gm * lda + k0 + col8, dst);
            else        *(uint4*)dst = make_uint4(0u, 0u, 0u, 0u);
        }
        {                                          // Bt tile: 64 x 32 direct rows
            int row  = tid >> 2;
            int col8 = (tid & 3) * 8;
            stage8(Bbase + (long long)(tileN + row) * ldb + k0 + col8,
                   &Bs[buf][row * LDSP + col8]);
        }
    };

    f32x8 acc[AM][AN];
#pragma unroll
    for (int i = 0; i < AM; ++i)
#pragma unroll
        for (int j = 0; j < AN; ++j)
#pragma unroll
            for (int r = 0; r < 8; ++r) acc[i][j][r] = 0.0f;

    const int nk = K >> 5;
    stage(0, 0);
    async_wait0();
    __syncthreads();

    for (int it = 0; it < nk; ++it) {
        const int cur = it & 1;
        if (it + 1 < nk) stage(cur ^ 1, (it + 1) * 32);   // async prefetch next chunk

        FragU afr[AM], bfr[AN];
#pragma unroll
        for (int i = 0; i < AM; ++i) load_fragA(afr[i], As[cur], waveM + i * 16, lane);
#pragma unroll
        for (int j = 0; j < AN; ++j) load_fragB(bfr[j], Bs[cur], waveN + j * 16, lane);
#pragma unroll
        for (int i = 0; i < AM; ++i)
#pragma unroll
            for (int j = 0; j < AN; ++j)
                acc[i][j] = wmma_bf16(afr[i], bfr[j], acc[i][j]);

        async_wait0();
        __syncthreads();
    }

    // epilogue per C/D layout: VGPR r -> M = r + 8*(lane>=16), N = lane&15
#pragma unroll
    for (int i = 0; i < AM; ++i) {
        int mbase = tileM + waveM + i * 16 + ((lane < 16) ? 0 : 8);
#pragma unroll
        for (int j = 0; j < AN; ++j) {
            int ncol = tileN + waveN + j * 16 + (lane & 15);
#pragma unroll
            for (int r = 0; r < 8; ++r) {
                int gm = mbase + r;
                if (gm < M) {
                    float v = acc[i][j][r];
                    if (bias)     v += bias[ncol];
                    if (residual) v += residual[(long long)gm * ldc + ncol];
                    C[(long long)gm * ldc + ncol] = v;
                    if (Cbf) Cbf[(long long)gm * ldc + ncol] = f32_to_bf16b(v);
                }
            }
        }
    }
}

// ---------------- batched A^T * B : C[b][m][n] = sum_k A[b][k][m] * B[b][k][n] ----------------
// 64x64 tile per block; blockIdx = (tile, batch, ksplit). bf16 out written TRANSPOSED [n][m]
// so downstream GEMMs can use it as a pre-transposed B operand.
__global__ __launch_bounds__(256) void gemm_atb_wmma(
    const unsigned short* __restrict__ A, int lda, long long aBatch,
    const unsigned short* __restrict__ B, int ldb, long long bBatch,
    float* __restrict__ Cpart, unsigned short* __restrict__ Cbf,
    int M, int N, int K, int kChunk)
{
    __shared__ unsigned short As[64 * LDSP];
    __shared__ unsigned short Bs[64 * LDSP];

    int tilesN = (N + 63) / 64;
    int tm = (blockIdx.x / tilesN) * 64;
    int tn = (blockIdx.x % tilesN) * 64;
    int b  = blockIdx.y;
    int z  = blockIdx.z;

    const unsigned short* Ab = A + (long long)b * aBatch;
    const unsigned short* Bb = B + (long long)b * bBatch;
    long long outOff = ((long long)z * gridDim.y + b) * ((long long)M * N);

    int tid = threadIdx.x, wid = tid >> 5, lane = tid & 31;
    int waveM = (wid % 2) * 32;
    int waveN = (wid / 2) * 16;

    f32x8 acc[2];
#pragma unroll
    for (int i = 0; i < 2; ++i)
#pragma unroll
        for (int r = 0; r < 8; ++r) acc[i][r] = 0.0f;

    int kBeg = z * kChunk;
    int kEnd = (kBeg + kChunk < K) ? (kBeg + kChunk) : K;

    for (int k0 = kBeg; k0 < kEnd; k0 += 32) {
        __syncthreads();
        int kb   = tid >> 3;
        int csub = (tid & 7) * 8;
        int gk   = k0 + kb;
        {   // A chunk transposed into As[m][k]
            U4 v; v.q = make_uint4(0u, 0u, 0u, 0u);
            if (gk < kEnd) v.q = *(const uint4*)(Ab + (long long)gk * lda + tm + csub);
#pragma unroll
            for (int i = 0; i < 8; ++i) As[(csub + i) * LDSP + kb] = v.s[i];
        }
        {   // B chunk transposed into Bs[n][k]
            U4 v; v.q = make_uint4(0u, 0u, 0u, 0u);
            if (gk < kEnd) v.q = *(const uint4*)(Bb + (long long)gk * ldb + tn + csub);
#pragma unroll
            for (int i = 0; i < 8; ++i) Bs[(csub + i) * LDSP + kb] = v.s[i];
        }
        if (k0 + 32 < kEnd)   // hint next chunk into cache (global_prefetch_b8)
            __builtin_prefetch(Ab + (long long)(k0 + 32 + (tid >> 3)) * lda + tm, 0, 0);
        __syncthreads();

        FragU a0, a1, b0;
        load_fragA(a0, As, waveM, lane);
        load_fragA(a1, As, waveM + 16, lane);
        load_fragB(b0, Bs, waveN, lane);
        acc[0] = wmma_bf16(a0, b0, acc[0]);
        acc[1] = wmma_bf16(a1, b0, acc[1]);
    }

#pragma unroll
    for (int i = 0; i < 2; ++i) {
        int mbase = tm + waveM + i * 16 + ((lane < 16) ? 0 : 8);
        int ncol  = tn + waveN + (lane & 15);
#pragma unroll
        for (int r = 0; r < 8; ++r) {
            float v = acc[i][r];
            Cpart[outOff + (long long)(mbase + r) * N + ncol] = v;
            if (Cbf)   // transposed bf16: [b][n][m]
                Cbf[(long long)b * M * N + (long long)ncol * M + (mbase + r)] = f32_to_bf16b(v);
        }
    }
}

// ---------------- elementwise / normalization kernels ----------------
__global__ __launch_bounds__(128) void ln_bf16_kernel(
    const float* __restrict__ x, const float* __restrict__ g, const float* __restrict__ bta,
    unsigned short* __restrict__ out, int act)
{
    __shared__ float red[128];
    long long row = blockIdx.x;
    int c = threadIdx.x;
    float v = x[row * 128 + c];
    red[c] = v; __syncthreads();
    for (int s = 64; s > 0; s >>= 1) { if (c < s) red[c] += red[c + s]; __syncthreads(); }
    float mean = red[0] * (1.0f / 128.0f); __syncthreads();
    float d = v - mean;
    red[c] = d * d; __syncthreads();
    for (int s = 64; s > 0; s >>= 1) { if (c < s) red[c] += red[c + s]; __syncthreads(); }
    float var = red[0] * (1.0f / 128.0f);
    float y = d * rsqrtf(var + 1e-5f) * g[c] + bta[c];
    if (act) y = gelu_f(y);
    out[row * 128 + c] = f32_to_bf16b(y);
}

__global__ __launch_bounds__(128) void softmax_row_bf16(
    const float* __restrict__ in, unsigned short* __restrict__ out)
{
    __shared__ float red[128];
    long long row = blockIdx.x;
    int c = threadIdx.x;
    float v = in[row * 128 + c];
    red[c] = v; __syncthreads();
    for (int s = 64; s > 0; s >>= 1) { if (c < s) red[c] = fmaxf(red[c], red[c + s]); __syncthreads(); }
    float mx = red[0]; __syncthreads();
    float e = expf(v - mx);
    red[c] = e; __syncthreads();
    for (int s = 64; s > 0; s >>= 1) { if (c < s) red[c] += red[c + s]; __syncthreads(); }
    out[row * 128 + c] = f32_to_bf16b(e / red[0]);
}

__global__ __launch_bounds__(256) void softmax_col_bf16(
    const float* __restrict__ in, long long inBStride, int inPitch, int inColOff,
    unsigned short* __restrict__ out, long long outBStride, int outPitch, int rows)
{
    __shared__ float red[256];
    int c = blockIdx.x, b = blockIdx.y, t = threadIdx.x;
    const float* p = in + (long long)b * inBStride + inColOff + c;
    float mx = -3.0e38f;
    for (int r = t; r < rows; r += 256) mx = fmaxf(mx, p[(long long)r * inPitch]);
    red[t] = mx; __syncthreads();
    for (int s = 128; s > 0; s >>= 1) { if (t < s) red[t] = fmaxf(red[t], red[t + s]); __syncthreads(); }
    mx = red[0]; __syncthreads();
    float sum = 0.0f;
    for (int r = t; r < rows; r += 256) sum += expf(p[(long long)r * inPitch] - mx);
    red[t] = sum; __syncthreads();
    for (int s = 128; s > 0; s >>= 1) { if (t < s) red[t] += red[t + s]; __syncthreads(); }
    float inv = 1.0f / red[0];
    unsigned short* q = out + (long long)b * outBStride + c;
    for (int r = t; r < rows; r += 256)
        q[(long long)r * outPitch] = f32_to_bf16b(expf(p[(long long)r * inPitch] - mx) * inv);
}

// v + depthwise3x3(v) + bias  -> bf16 (channel-last grid HxW)
__global__ __launch_bounds__(256) void dw_add_bf16(
    const float* __restrict__ vin, long long bStride, int pitch, int colOff,
    const float* __restrict__ w, const float* __restrict__ bias,
    unsigned short* __restrict__ out, long long outBStride, int outPitch,
    int Hh, int Ww, int CH, long long total)
{
    long long idx = (long long)blockIdx.x * blockDim.x + threadIdx.x;
    if (idx >= total) return;
    int c = (int)(idx % CH);
    long long r = idx / CH;
    int n = (int)(r % (Hh * Ww));
    int b = (int)(r / (Hh * Ww));
    int y = n / Ww, x = n % Ww;
    const float* base = vin + (long long)b * bStride + colOff + c;
    float acc = bias[c];
#pragma unroll
    for (int ky = 0; ky < 3; ++ky)
#pragma unroll
        for (int kx = 0; kx < 3; ++kx) {
            int yy = y + ky - 1, xx = x + kx - 1;
            if (yy >= 0 && yy < Hh && xx >= 0 && xx < Ww)
                acc += base[(long long)(yy * Ww + xx) * pitch] * w[c * 9 + ky * 3 + kx];
        }
    float center = base[(long long)n * pitch];
    out[(long long)b * outBStride + (long long)n * outPitch + c] = f32_to_bf16b(center + acc);
}

// gelu(depthwise3x3(in) + bias) -> bf16   (mixffn dwconv, channel-last)
__global__ __launch_bounds__(256) void dwc_gelu_bf16(
    const float* __restrict__ in, const float* __restrict__ w, const float* __restrict__ bias,
    unsigned short* __restrict__ out, int Hh, int Ww, int CH, long long total)
{
    long long idx = (long long)blockIdx.x * blockDim.x + threadIdx.x;
    if (idx >= total) return;
    int c = (int)(idx % CH);
    long long r = idx / CH;
    int n = (int)(r % (Hh * Ww));
    int b = (int)(r / (Hh * Ww));
    int y = n / Ww, x = n % Ww;
    const float* base = in + ((long long)b * (Hh * Ww)) * CH + c;
    float acc = bias[c];
#pragma unroll
    for (int ky = 0; ky < 3; ++ky)
#pragma unroll
        for (int kx = 0; kx < 3; ++kx) {
            int yy = y + ky - 1, xx = x + kx - 1;
            if (yy >= 0 && yy < Hh && xx >= 0 && xx < Ww)
                acc += base[(long long)(yy * Ww + xx) * CH] * w[c * 9 + ky * 3 + kx];
        }
    out[idx] = f32_to_bf16b(gelu_f(acc));
}

// DES stage 1: y[t][kd][:] = x_row(8x16) @ pr_w + pr_b
__global__ __launch_bounds__(256) void des_stage1(
    const float* __restrict__ x, const float* __restrict__ pr_w, const float* __restrict__ pr_b,
    float* __restrict__ y, long long total)
{
    long long idx = (long long)blockIdx.x * blockDim.x + threadIdx.x;
    if (idx >= total) return;
    int kd = (int)(idx & 7);
    long long t = idx >> 3;
    int b = (int)(t / 3136);
    int m = (int)(t % 3136);
    const float* xb = x + (long long)b * 3136 * 128;
    float xi[16];
#pragma unroll
    for (int i = 0; i < 16; ++i) xi[i] = xb[(long long)(kd * 16 + i) * 3136 + m];
#pragma unroll
    for (int j = 0; j < 16; ++j) {
        float s = pr_b[j];
#pragma unroll
        for (int i = 0; i < 16; ++i) s += xi[i] * pr_w[i * 16 + j];
        y[idx * 16 + j] = s;
    }
}

// DES stage 2: gelu-transpose, 8x8 matmul, transpose back
__global__ __launch_bounds__(256) void des_stage2(
    const float* __restrict__ y, const float* __restrict__ pl_w, const float* __restrict__ pl_b,
    float* __restrict__ shortcut, long long total)
{
    long long idx = (long long)blockIdx.x * blockDim.x + threadIdx.x;
    if (idx >= total) return;
    int p = (int)(idx & 15);
    long long t = idx >> 4;
    float z[8];
#pragma unroll
    for (int kd = 0; kd < 8; ++kd) z[kd] = gelu_f(y[(t * 8 + kd) * 16 + p]);
#pragma unroll
    for (int kq = 0; kq < 8; ++kq) {
        float s = pl_b[kq];
#pragma unroll
        for (int kd = 0; kd < 8; ++kd) s += z[kd] * pl_w[kd * 8 + kq];
        shortcut[t * 128 + kq * 16 + p] = s;
    }
}

__global__ __launch_bounds__(256) void combine_tx(
    const float* __restrict__ x, const float* __restrict__ eff, const float* __restrict__ eff2,
    const float* __restrict__ sc, const float* __restrict__ dw3,
    float* __restrict__ tx, long long total)
{
    long long idx = (long long)blockIdx.x * blockDim.x + threadIdx.x;
    if (idx >= total) return;
    int c = (int)(idx & 127);
    tx[idx] = x[idx] + dw3[c * 2 + 0] * eff[idx] + dw3[c * 2 + 1] * eff2[idx] + sc[idx];
}

// sum over K-splits, write bf16 TRANSPOSED per batch: out[b][n][m]
__global__ __launch_bounds__(256) void reduce_splits_bf16_t(
    const float* __restrict__ part, int nsplit, long long sliceElems,
    unsigned short* __restrict__ out, int Mm, int Nn2, long long total)
{
    long long idx = (long long)blockIdx.x * blockDim.x + threadIdx.x;
    if (idx >= total) return;
    float s = 0.0f;
    for (int i = 0; i < nsplit; ++i) s += part[(long long)i * sliceElems + idx];
    int n = (int)(idx % Nn2);
    long long r = idx / Nn2;
    int m = (int)(r % Mm);
    int b = (int)(r / Mm);
    out[(long long)b * Mm * Nn2 + (long long)n * Mm + m] = f32_to_bf16b(s);
}

__global__ __launch_bounds__(256) void w_to_bf16(
    const float* __restrict__ w, unsigned short* __restrict__ out, long long n)
{
    long long i = (long long)blockIdx.x * blockDim.x + threadIdx.x;
    if (i < n) out[i] = f32_to_bf16b(w[i]);
}

// W [R,Cw] row-major -> bf16 [Cw,R] (used to make [N,K] weight layouts from [K,N])
__global__ __launch_bounds__(256) void w_to_bf16_t(
    const float* __restrict__ w, unsigned short* __restrict__ out, int R, int Cw)
{
    long long i = (long long)blockIdx.x * blockDim.x + threadIdx.x;
    if (i >= (long long)R * Cw) return;
    int r = (int)(i / Cw);
    int c = (int)(i % Cw);
    out[(long long)c * R + r] = f32_to_bf16b(w[i]);
}

// im2col for stride==kernel patch conv -> bf16 rows [b*OHW*OHW, 128*S*S], k = c*S*S + ky*S + kx
__global__ __launch_bounds__(256) void im2col_bf16(
    const unsigned short* __restrict__ xn, unsigned short* __restrict__ out,
    int S, int OHW, long long total)
{
    long long idx = (long long)blockIdx.x * blockDim.x + threadIdx.x;
    if (idx >= total) return;
    int kpitch = 128 * S * S;
    int kk = (int)(idx % kpitch);
    long long r = idx / kpitch;
    int pos = (int)(r % (OHW * OHW));
    int b = (int)(r / (OHW * OHW));
    int c = kk / (S * S);
    int rem = kk % (S * S);
    int ky = rem / S, kx = rem % S;
    int oy = pos / OHW, ox = pos % OHW;
    int n = (oy * S + ky) * 56 + ox * S + kx;
    out[idx] = xn[((long long)b * 3136 + n) * 128 + c];
}

// ---------------- launcher ----------------
static inline long long cdivll(long long a, long long b) { return (a + b - 1) / b; }

extern "C" void kernel_launch(void* const* d_in, const int* in_sizes, int n_in,
                              void* d_out, int out_size, void* d_ws, size_t ws_size,
                              hipStream_t stream) {
    const int Bn = 32, Nn = 3136, Cc = 128, Mfull = Bn * Nn, DFFc = 512;

    const float* x        = (const float*)d_in[0];
    const float* ln1_g    = (const float*)d_in[1];
    const float* ln1_b    = (const float*)d_in[2];
    const float* q_w      = (const float*)d_in[3];
    const float* q_b      = (const float*)d_in[4];
    const float* keys_w   = (const float*)d_in[5];
    const float* keys_b   = (const float*)d_in[6];
    const float* values_w = (const float*)d_in[7];
    const float* values_b = (const float*)d_in[8];
    const float* sr1_w    = (const float*)d_in[9];
    const float* sr1_b    = (const float*)d_in[10];
    const float* an1_g    = (const float*)d_in[11];
    const float* an1_b    = (const float*)d_in[12];
    const float* sr2_w    = (const float*)d_in[13];
    const float* sr2_b    = (const float*)d_in[14];
    const float* an2_g    = (const float*)d_in[15];
    const float* an2_b    = (const float*)d_in[16];
    const float* kv1_w    = (const float*)d_in[17];
    const float* kv1_b    = (const float*)d_in[18];
    const float* kv2_w    = (const float*)d_in[19];
    const float* kv2_b    = (const float*)d_in[20];
    const float* lc1_w    = (const float*)d_in[21];
    const float* lc1_b    = (const float*)d_in[22];
    const float* lc2_w    = (const float*)d_in[23];
    const float* lc2_b    = (const float*)d_in[24];
    const float* rp_w     = (const float*)d_in[25];
    const float* rp_b     = (const float*)d_in[26];
    const float* rp12_w   = (const float*)d_in[27];
    const float* rp12_b   = (const float*)d_in[28];
    const float* dw3_w    = (const float*)d_in[29];
    const float* pr_w     = (const float*)d_in[30];
    const float* pr_b     = (const float*)d_in[31];
    const float* pl_w     = (const float*)d_in[32];
    const float* pl_b     = (const float*)d_in[33];
    const float* ln2_g    = (const float*)d_in[34];
    const float* ln2_b    = (const float*)d_in[35];
    const float* fc1_w    = (const float*)d_in[36];
    const float* fc1_b    = (const float*)d_in[37];
    const float* dwc_w    = (const float*)d_in[38];
    const float* dwc_b    = (const float*)d_in[39];
    const float* fc2_w    = (const float*)d_in[40];
    const float* fc2_b    = (const float*)d_in[41];

    char* ws = (char*)d_ws;
    size_t off = 0;
    auto alloc = [&](size_t bytes) -> void* {
        off = (off + 255) & ~(size_t)255;
        void* p = ws + off;
        off += bytes;
        return p;
    };

    const long long ACT = (long long)Mfull * Cc;
    // bf16 activations
    unsigned short* xn_bf     = (unsigned short*)alloc(ACT * 2);
    unsigned short* qs_bf     = (unsigned short*)alloc(ACT * 2);
    unsigned short* keys_bf   = (unsigned short*)alloc(ACT * 2);
    unsigned short* vals_bf   = (unsigned short*)alloc(ACT * 2);
    unsigned short* im2col    = (unsigned short*)alloc(ACT * 2);   // reused S=8 / S=4
    unsigned short* avcat_bf  = (unsigned short*)alloc(ACT * 2);
    unsigned short* av_bf     = (unsigned short*)alloc(ACT * 2);
    unsigned short* txln_bf   = (unsigned short*)alloc(ACT * 2);
    unsigned short* dwcact_bf = (unsigned short*)alloc((long long)Mfull * DFFc * 2);
    unsigned short* x1_bf     = (unsigned short*)alloc(1568LL * 128 * 2);
    unsigned short* x2_bf     = (unsigned short*)alloc(6272LL * 128 * 2);
    unsigned short* k1_bf     = (unsigned short*)alloc(1568LL * 64 * 2);
    unsigned short* v1_bf     = (unsigned short*)alloc(1568LL * 64 * 2);
    unsigned short* k2_bf     = (unsigned short*)alloc(6272LL * 64 * 2);
    unsigned short* v2_bf     = (unsigned short*)alloc(6272LL * 64 * 2);
    unsigned short* ctx1_bf   = (unsigned short*)alloc(32LL * 64 * 64 * 2);   // [b][n][m]
    unsigned short* ctx2_bf   = (unsigned short*)alloc(32LL * 64 * 64 * 2);   // [b][n][m]
    unsigned short* ctx_bf    = (unsigned short*)alloc(32LL * 128 * 128 * 2); // [b][n][m]
    // bf16 weights, all stored [N,K]
    unsigned short* qw_bf    = (unsigned short*)alloc(16384 * 2);
    unsigned short* keysw_bf = (unsigned short*)alloc(16384 * 2);
    unsigned short* valsw_bf = (unsigned short*)alloc(16384 * 2);
    unsigned short* kv1w_bf  = (unsigned short*)alloc(16384 * 2);
    unsigned short* kv2w_bf  = (unsigned short*)alloc(16384 * 2);
    unsigned short* fc1w_bf  = (unsigned short*)alloc(65536 * 2);
    unsigned short* fc2w_bf  = (unsigned short*)alloc(65536 * 2);
    unsigned short* rpw_bf   = (unsigned short*)alloc(16384 * 2);
    unsigned short* rp12w_bf = (unsigned short*)alloc(16384 * 2);
    unsigned short* sr1w_bf  = (unsigned short*)alloc(128LL * 8192 * 2);
    unsigned short* sr2w_bf  = (unsigned short*)alloc(128LL * 2048 * 2);
    // f32 scratch
    float* F1 = (float*)alloc(ACT * 4);     // q_pre -> vals f32 -> av f32
    float* F2 = (float*)alloc(ACT * 4);     // keys_pre -> avcat f32
    float* F3 = (float*)alloc(ACT * 4);     // eff2
    float* F4 = (float*)alloc(ACT * 4);     // eff
    float* F5 = (float*)alloc(ACT * 4);     // y_des
    float* F6 = (float*)alloc(ACT * 4);     // shortcut
    float* F7 = (float*)alloc(ACT * 4);     // tx
    float* fc1out   = (float*)alloc((long long)Mfull * DFFc * 4);
    float* x1pre    = (float*)alloc(1568LL * 128 * 4);
    float* x2pre    = (float*)alloc(6272LL * 128 * 4);
    float* kv1f     = (float*)alloc(1568LL * 128 * 4);
    float* kv2f     = (float*)alloc(6272LL * 128 * 4);
    float* ctx1f    = (float*)alloc(32LL * 64 * 64 * 4);
    float* ctx2f    = (float*)alloc(32LL * 64 * 64 * 4);
    float* ctxpart  = (float*)alloc(8LL * 32 * 128 * 128 * 4);
    (void)ws_size; (void)in_sizes; (void)n_in; (void)out_size;

    auto gemmA = [&](const unsigned short* A, int lda, const unsigned short* Bm, int ldb,
                     float* Cf, unsigned short* Cb, int ldc, const float* bias,
                     const float* resid, int M_, int N_, int K_) {
        dim3 g((unsigned)cdivll(M_, 128), (unsigned)(N_ / 64));
        gemm_bf16_wmma<128, 4, 2><<<g, dim3(256), 0, stream>>>(
            A, lda, Bm, ldb, 0LL, 1, Cf, Cb, ldc, bias, resid, M_, N_, K_);
    };
    auto gemmB = [&](const unsigned short* A, int lda, const unsigned short* Bm, int ldb,
                     long long bStride, float* Cf, unsigned short* Cb, int ldc,
                     int M_, int N_, int K_) {
        dim3 g((unsigned)cdivll(M_, 64), (unsigned)(N_ / 64));
        gemm_bf16_wmma<64, 2, 4><<<g, dim3(256), 0, stream>>>(
            A, lda, Bm, ldb, bStride, Nn, Cf, Cb, ldc, nullptr, nullptr, M_, N_, K_);
    };

    // 0) weight conversions: [K,N] weights transposed to [N,K]; OIHW 1x1/patch weights are already [N,K]
    w_to_bf16_t<<<64, 256, 0, stream>>>(q_w, qw_bf, 128, 128);
    w_to_bf16_t<<<64, 256, 0, stream>>>(keys_w, keysw_bf, 128, 128);
    w_to_bf16_t<<<64, 256, 0, stream>>>(values_w, valsw_bf, 128, 128);
    w_to_bf16_t<<<64, 256, 0, stream>>>(kv1_w, kv1w_bf, 128, 128);
    w_to_bf16_t<<<64, 256, 0, stream>>>(kv2_w, kv2w_bf, 128, 128);
    w_to_bf16_t<<<256, 256, 0, stream>>>(fc1_w, fc1w_bf, 128, 512);
    w_to_bf16_t<<<256, 256, 0, stream>>>(fc2_w, fc2w_bf, 512, 128);
    w_to_bf16<<<64, 256, 0, stream>>>(rp_w, rpw_bf, 16384);
    w_to_bf16<<<64, 256, 0, stream>>>(rp12_w, rp12w_bf, 16384);
    w_to_bf16<<<4096, 256, 0, stream>>>(sr1_w, sr1w_bf, 128LL * 8192);
    w_to_bf16<<<1024, 256, 0, stream>>>(sr2_w, sr2w_bf, 128LL * 2048);

    // 1) LN1 -> xn (bf16)
    ln_bf16_kernel<<<Mfull, 128, 0, stream>>>(x, ln1_g, ln1_b, xn_bf, 0);

    // 2) q / keys / values projections
    gemmA(xn_bf, 128, qw_bf, 128, F1, nullptr, 128, q_b, nullptr, Mfull, 128, 128);
    softmax_row_bf16<<<Mfull, 128, 0, stream>>>(F1, qs_bf);
    gemmA(xn_bf, 128, keysw_bf, 128, F2, nullptr, 128, keys_b, nullptr, Mfull, 128, 128);
    softmax_col_bf16<<<dim3(128, 32), 256, 0, stream>>>(
        F2, (long long)Nn * 128, 128, 0, keys_bf, (long long)Nn * 128, 128, Nn);
    gemmA(xn_bf, 128, valsw_bf, 128, F1, vals_bf, 128, values_b, nullptr, Mfull, 128, 128);

    // 3) sr1 / sr2 patch convs via im2col GEMM, then LN+GELU
    im2col_bf16<<<(unsigned)cdivll(ACT, 256), 256, 0, stream>>>(xn_bf, im2col, 8, 7, ACT);
    gemmA(im2col, 8192, sr1w_bf, 8192, x1pre, nullptr, 128, sr1_b, nullptr, 1568, 128, 8192);
    ln_bf16_kernel<<<1568, 128, 0, stream>>>(x1pre, an1_g, an1_b, x1_bf, 1);
    im2col_bf16<<<(unsigned)cdivll(ACT, 256), 256, 0, stream>>>(xn_bf, im2col, 4, 14, ACT);
    gemmA(im2col, 2048, sr2w_bf, 2048, x2pre, nullptr, 128, sr2_b, nullptr, 6272, 128, 2048);
    ln_bf16_kernel<<<6272, 128, 0, stream>>>(x2pre, an2_g, an2_b, x2_bf, 1);

    // 4) kv projections, k softmax over tokens, v + local depthwise conv
    gemmA(x1_bf, 128, kv1w_bf, 128, kv1f, nullptr, 128, kv1_b, nullptr, 1568, 128, 128);
    softmax_col_bf16<<<dim3(64, 32), 256, 0, stream>>>(
        kv1f, 49LL * 128, 128, 0, k1_bf, 49LL * 64, 64, 49);
    dw_add_bf16<<<(unsigned)cdivll(32LL * 49 * 64, 256), 256, 0, stream>>>(
        kv1f, 49LL * 128, 128, 64, lc1_w, lc1_b, v1_bf, 49LL * 64, 64, 7, 7, 64, 32LL * 49 * 64);
    gemmA(x2_bf, 128, kv2w_bf, 128, kv2f, nullptr, 128, kv2_b, nullptr, 6272, 128, 128);
    softmax_col_bf16<<<dim3(64, 32), 256, 0, stream>>>(
        kv2f, 196LL * 128, 128, 0, k2_bf, 196LL * 64, 64, 196);
    dw_add_bf16<<<(unsigned)cdivll(32LL * 196 * 64, 256), 256, 0, stream>>>(
        kv2f, 196LL * 128, 128, 64, lc2_w, lc2_b, v2_bf, 196LL * 64, 64, 14, 14, 64, 32LL * 196 * 64);

    // 5) context matrices (bf16 outputs stored transposed [b][n][m] for GEMM-B reuse)
    gemm_atb_wmma<<<dim3(1, 32, 1), 256, 0, stream>>>(
        k1_bf, 64, 49LL * 64, v1_bf, 64, 49LL * 64, ctx1f, ctx1_bf, 64, 64, 49, 49);
    gemm_atb_wmma<<<dim3(1, 32, 1), 256, 0, stream>>>(
        k2_bf, 64, 196LL * 64, v2_bf, 64, 196LL * 64, ctx2f, ctx2_bf, 64, 64, 196, 196);
    gemm_atb_wmma<<<dim3(4, 32, 8), 256, 0, stream>>>(
        keys_bf, 128, (long long)Nn * 128, vals_bf, 128, (long long)Nn * 128,
        ctxpart, nullptr, 128, 128, Nn, 392);
    reduce_splits_bf16_t<<<(unsigned)cdivll(32LL * 128 * 128, 256), 256, 0, stream>>>(
        ctxpart, 8, 32LL * 128 * 128, ctx_bf, 128, 128, 32LL * 128 * 128);

    // 6) attention-value products (per-batch pre-transposed B)
    gemmB(qs_bf, 128, ctx1_bf, 64, 64LL * 64, F2, avcat_bf, 128, Mfull, 64, 64);
    gemmB(qs_bf + 64, 128, ctx2_bf, 64, 64LL * 64, F2 + 64, avcat_bf + 64, 128, Mfull, 64, 64);
    gemmB(qs_bf, 128, ctx_bf, 128, 128LL * 128, F1, av_bf, 128, Mfull, 128, 128);

    // 7) reprojection 1x1 convs
    gemmA(avcat_bf, 128, rp12w_bf, 128, F3, nullptr, 128, rp12_b, nullptr, Mfull, 128, 128); // eff2
    gemmA(av_bf, 128, rpw_bf, 128, F4, nullptr, 128, rp_b, nullptr, Mfull, 128, 128);        // eff

    // 8) DES shortcut
    des_stage1<<<(unsigned)cdivll((long long)Mfull * 8, 256), 256, 0, stream>>>(
        x, pr_w, pr_b, F5, (long long)Mfull * 8);
    des_stage2<<<(unsigned)cdivll((long long)Mfull * 16, 256), 256, 0, stream>>>(
        F5, pl_w, pl_b, F6, (long long)Mfull * 16);

    // 9) combine + LN2
    combine_tx<<<(unsigned)cdivll(ACT, 256), 256, 0, stream>>>(x, F4, F3, F6, dw3_w, F7, ACT);
    ln_bf16_kernel<<<Mfull, 128, 0, stream>>>(F7, ln2_g, ln2_b, txln_bf, 0);

    // 10) MixFFN: fc1 -> depthwise+gelu -> fc2 (+residual tx) -> d_out
    gemmA(txln_bf, 128, fc1w_bf, 128, fc1out, nullptr, 512, fc1_b, nullptr, Mfull, 512, 128);
    dwc_gelu_bf16<<<(unsigned)cdivll((long long)Mfull * DFFc, 256), 256, 0, stream>>>(
        fc1out, dwc_w, dwc_b, dwcact_bf, 56, 56, 512, (long long)Mfull * DFFc);
    gemmA(dwcact_bf, 512, fc2w_bf, 512, (float*)d_out, nullptr, 128, fc2_b, F7, Mfull, 128, 512);
}
